// GCN_85014582657655
// MI455X (gfx1250) — matrix operand order repaired and checked by reading, not compiled, not run on hardware
//
#include <hip/hip_runtime.h>
#include <hip/hip_bf16.h>

#define N_NODES 20000
#define N_EDGES 512000
#define N_GRAPHS 64
#define LN_EPS 1e-5f

typedef __bf16 bf16;
typedef __attribute__((ext_vector_type(16))) __bf16 v16bf;
typedef __attribute__((ext_vector_type(8)))  __bf16 bf16x8;
typedef __attribute__((ext_vector_type(2)))  __bf16 bf16x2;
typedef __attribute__((ext_vector_type(8)))  float  v8f;
typedef __attribute__((ext_vector_type(8)))  float  f32x8;
typedef __attribute__((ext_vector_type(2)))  float  f32x2;

union AF { v16bf v; bf16x8 h[2]; };

// native f32->bf16 (lowers to v_cvt_pk_bf16_f32)
__device__ __forceinline__ bf16 f2bf(float f) { return (bf16)f; }
__device__ __forceinline__ bf16x2 f2bf2(float a, float b) {
  f32x2 v = {a, b};
  return __builtin_convertvector(v, bf16x2);
}
__device__ __forceinline__ bf16x8 f2bf8(f32x8 v) {
  return __builtin_convertvector(v, bf16x8);
}

// Weight W[128][128] f32 (row-major) -> LDS swizzled [k/16][n][k%16] bf16 (32KB)
__device__ __forceinline__ void load_wlds(bf16* wl, const float* W) {
  for (int idx = threadIdx.x; idx < 128 * 128; idx += blockDim.x) {
    int k = idx >> 7, n = idx & 127;
    wl[((k >> 4) << 11) + (n << 4) + (k & 15)] = f2bf(W[idx]);
  }
}

// B fragment: one N column, K = 8h+0..7 and 16+8h+0..7 (table order) -> two ds_load_b128
__device__ __forceinline__ v16bf bfrag(const bf16* wl, int kg, int ncol) {
  AF b;
  const bf16* p = wl + ((kg & ~1) << 11) + (ncol << 4) + ((kg & 1) << 3);
  b.h[0] = *(const bf16x8*)p;
  b.h[1] = *(const bf16x8*)(p + 2048);
  return b.v;
}

// A fragment per ISA 16-bit A 16x32 layout. rowp = rowbase + (lane>>4)*8
__device__ __forceinline__ v16bf afrag_p(const bf16* rowp, int ks) {
  AF a;
  const bf16* p = rowp + ks * 32;
  a.h[0] = *(const bf16x8*)p;        // K = ks*32 + koff + 0..7
  a.h[1] = *(const bf16x8*)(p + 16); // K = ks*32 + 16 + koff + 0..7
  return a.v;
}

// 16x128 tile, K=128.  B fragments for a whole k-step are fetched up-front so the
// 16 ds_load_b128 can issue as a clause and overlap the 8 WMMAs (partial dscnt waits).
__device__ __forceinline__ void gemm128(v8f* acc, const bf16* rowp, const bf16* wl, int lane) {
  int n0 = lane & 15, kh = lane >> 4;
#pragma unroll
  for (int ks = 0; ks < 4; ++ks) {
    v16bf a = afrag_p(rowp, ks);
    int kg = ks * 2 + kh;
    v16bf bfr[8];
#pragma unroll
    for (int t = 0; t < 8; ++t) bfr[t] = bfrag(wl, kg, t * 16 + n0);
#pragma unroll
    for (int t = 0; t < 8; ++t)
      acc[t] = __builtin_amdgcn_wmma_f32_16x16x32_bf16(false, a, false, bfr[t], (short)0,
                                                       acc[t], false, false);
  }
}

// acc (+bias, optional relu) -> bf16 stage tile [16][136]; packed f32->bf16 converts
__device__ __forceinline__ void epi_stage(bf16* st, const v8f* acc, const float* bias,
                                          int lane, bool relu) {
  int n0 = lane & 15, kh = lane >> 4;
#pragma unroll
  for (int t = 0; t < 8; ++t) {
    int col = t * 16 + n0;
    float bv = bias[col];
#pragma unroll
    for (int r = 0; r < 8; r += 2) {
      float v0 = acc[t][r] + bv;
      float v1 = acc[t][r + 1] + bv;
      if (relu) { v0 = v0 > 0.f ? v0 : 0.f; v1 = v1 > 0.f ? v1 : 0.f; }
      bf16x2 p = f2bf2(v0, v1);
      st[(r + 0 + kh * 8) * 136 + col] = p.x;
      st[(r + 1 + kh * 8) * 136 + col] = p.y;
    }
  }
}

// coalesced 16B stores of a 16x128 bf16 stage tile
__device__ __forceinline__ void store_tile(bf16* out, long ldo, long rowbase, long rowmax,
                                           const bf16* st, int lane) {
#pragma unroll
  for (int i = 0; i < 8; ++i) {
    int chunk = lane + i * 32;
    int row = chunk >> 4, cc = chunk & 15;
    long gr = rowbase + row;
    if (gr < rowmax)
      *(uint4*)(out + gr * ldo + (long)cc * 8) = *(const uint4*)(st + row * 136 + cc * 8);
  }
}

// ---------------- Kernel 1: edge MLP  ea = relu(attr@W1+b1)@W2+b2 (bf16 out) ------------
__global__ __launch_bounds__(128) void edge_mlp_kernel(
    const float* __restrict__ eattr, const float* __restrict__ w1, const float* __restrict__ b1,
    const float* __restrict__ w2, const float* __restrict__ b2, bf16* __restrict__ out) {
  __shared__ __align__(16) bf16 w1l[2 * 2048];   // K padded 16->32, top half zero
  __shared__ __align__(16) bf16 w2l[8 * 2048];
  __shared__ __align__(16) bf16 stage[4][16 * 136];
  for (int idx = threadIdx.x; idx < 4096; idx += blockDim.x) {
    int kg = idx >> 11, rem = idx & 2047, n = rem >> 4, kk = rem & 15;
    int k = kg * 16 + kk;
    w1l[idx] = (k < 16) ? f2bf(w1[k * 128 + n]) : f2bf(0.f);
  }
  load_wlds(w2l, w2);
  __syncthreads();

  int lane = threadIdx.x & 31, wave = threadIdx.x >> 5;
  int n0 = lane & 15, kh = lane >> 4;
  bf16* st = stage[wave];
  const uint4 z4 = {0, 0, 0, 0};

  for (int it = 0; it < 4; ++it) {
    long ebase = ((long)blockIdx.x * 4 + it) * 64 + wave * 16;
    if (it < 3)
      __builtin_prefetch(eattr + (ebase + 64 + n0) * 16, 0, 1);
    // GEMM1: K=16 zero-padded to 32 (one WMMA per N-tile)
    AF a;
    a.h[0] = f2bf8(*(const f32x8*)(eattr + (ebase + n0) * 16 + kh * 8));
    a.h[1] = __builtin_bit_cast(bf16x8, z4);   // K >= 16 is zero padding
    v16bf bfr[8];
#pragma unroll
    for (int t = 0; t < 8; ++t) bfr[t] = bfrag(w1l, kh, t * 16 + n0);
    v8f acc[8] = {};
#pragma unroll
    for (int t = 0; t < 8; ++t)
      acc[t] = __builtin_amdgcn_wmma_f32_16x16x32_bf16(false, a.v, false, bfr[t], (short)0,
                                                       acc[t], false, false);
    epi_stage(st, acc, b1, lane, true);
    // GEMM2: K=128
    v8f acc2[8] = {};
    gemm128(acc2, st + n0 * 136 + kh * 8, w2l, lane);
    epi_stage(st, acc2, b2, lane, false);
    store_tile(out, 128, ebase, N_EDGES, st, lane);
  }
}

// ------------- Kernel 2: fused e=ea@lw+lb, m=relu(x[src]+e), scatter-add to aggr --------
__global__ __launch_bounds__(128) void branch_edge_kernel(
    const bf16* __restrict__ ea, const float* __restrict__ linw, const float* __restrict__ linb,
    const float* __restrict__ x, const int* __restrict__ src, const int* __restrict__ dst,
    float* __restrict__ aggr) {
  int br = blockIdx.y;
  __shared__ __align__(16) bf16 wl[8 * 2048];
  load_wlds(wl, linw + br * 16384);
  __syncthreads();
  int lane = threadIdx.x & 31, wave = threadIdx.x >> 5;
  int n0 = lane & 15, kh = lane >> 4;
  const float* lb = linb + br * 128;
  float* ag = aggr + (long)br * N_NODES * 128;

  for (int it = 0; it < 4; ++it) {
    long ebase = ((long)blockIdx.x * 4 + it) * 64 + wave * 16;
    if (it < 3)
      __builtin_prefetch(ea + (ebase + 64 + n0) * 128, 0, 1);  // global_prefetch_b8
    v8f acc[8] = {};
    gemm128(acc, ea + (ebase + n0) * 128 + kh * 8, wl, lane);
#pragma unroll
    for (int r = 0; r < 8; ++r) {
      long edge = ebase + r + kh * 8;
      int sn = src[edge], dn = dst[edge];
      const float* xs = x + (long)sn * 128;
      float* ad = ag + (long)dn * 128;
#pragma unroll
      for (int t = 0; t < 8; ++t) {
        int col = t * 16 + n0;
        float v = acc[t][r] + lb[col] + xs[col];
        v = v > 0.f ? v : 0.f;
        atomicAdd(ad + col, v);
      }
    }
  }
}

// --------------- Kernel 3: t1 = relu((x + aggr_k) @ W1_k + b1_k)  (bf16 out) ------------
__global__ __launch_bounds__(128) void node_mlp1_kernel(
    const float* __restrict__ x, const float* __restrict__ aggr,
    const float* __restrict__ w1, const float* __restrict__ b1, bf16* __restrict__ t1) {
  int br = blockIdx.y;
  __shared__ __align__(16) bf16 wl[8 * 2048];
  __shared__ __align__(16) bf16 stage[4][16 * 136];
  load_wlds(wl, w1 + br * 16384);
  __syncthreads();
  int lane = threadIdx.x & 31, wave = threadIdx.x >> 5;
  int n0 = lane & 15, kh = lane >> 4;
  long rbase = (long)blockIdx.x * 64 + wave * 16;
  bf16* st = stage[wave];
  const float* agb = aggr + (long)br * N_NODES * 128;

  // build h tile in bf16: 16 rows x 16 chunks of 8 cols, one ds_store_b128 per chunk
  for (int i = lane; i < 16 * 16; i += 32) {
    int row = i >> 4, c8 = (i & 15) * 8;
    long gr = rbase + row; if (gr > N_NODES - 1) gr = N_NODES - 1;
    f32x8 xv = *(const f32x8*)(x + gr * 128 + c8);
    f32x8 av = *(const f32x8*)(agb + gr * 128 + c8);
    *(bf16x8*)(st + row * 136 + c8) = f2bf8(xv + av);
  }
  v8f acc[8] = {};
  gemm128(acc, st + n0 * 136 + kh * 8, wl, lane);
  epi_stage(st, acc, b1 + br * 128, lane, true);
  store_tile(t1 + (long)br * N_NODES * 128, 128, rbase, N_NODES, st, lane);
}

// ------------- Kernel 4: xcat[:,br*128:] = relu(t1 @ W2_k + b2_k)  (bf16 out) -----------
__global__ __launch_bounds__(128) void node_mlp2_kernel(
    const bf16* __restrict__ t1, const float* __restrict__ w2, const float* __restrict__ b2,
    bf16* __restrict__ xcat) {
  int br = blockIdx.y;
  __shared__ __align__(16) bf16 wl[8 * 2048];
  __shared__ __align__(16) bf16 stage[4][16 * 136];
  load_wlds(wl, w2 + br * 16384);
  __syncthreads();
  int lane = threadIdx.x & 31, wave = threadIdx.x >> 5;
  int n0 = lane & 15, kh = lane >> 4;
  long rbase = (long)blockIdx.x * 64 + wave * 16;
  long rowg = rbase + n0; if (rowg > N_NODES - 1) rowg = N_NODES - 1;
  v8f acc[8] = {};
  gemm128(acc, t1 + ((long)br * N_NODES + rowg) * 128 + kh * 8, wl, lane);
  bf16* st = stage[wave];
  epi_stage(st, acc, b2 + br * 128, lane, true);
  store_tile(xcat + br * 128, 384, rbase, N_NODES, st, lane);
}

// ------- Kernel 5: x_out = LayerNorm(relu(xcat @ LW + lb)); K=384 in 3 LDS chunks -------
__global__ __launch_bounds__(128) void lin_ln_kernel(
    const bf16* __restrict__ xcat, const float* __restrict__ lw, const float* __restrict__ lb,
    const float* __restrict__ g, const float* __restrict__ bb, float* __restrict__ out) {
  __shared__ __align__(16) bf16 wl[8 * 2048];
  int lane = threadIdx.x & 31, wave = threadIdx.x >> 5;
  int n0 = lane & 15, kh = lane >> 4;
  long rbase = (long)blockIdx.x * 64 + wave * 16;
  long rowg = rbase + n0; if (rowg > N_NODES - 1) rowg = N_NODES - 1;
  v8f acc[8] = {};
  for (int kc = 0; kc < 3; ++kc) {
    __syncthreads();
    load_wlds(wl, lw + kc * 16384);
    __syncthreads();
    gemm128(acc, xcat + rowg * 384 + kc * 128 + kh * 8, wl, lane);
  }
  float s1[8] = {}, s2[8] = {};
#pragma unroll
  for (int t = 0; t < 8; ++t) {
    int col = t * 16 + n0;
    float bv = lb[col];
#pragma unroll
    for (int r = 0; r < 8; ++r) {
      float v = acc[t][r] + bv; v = v > 0.f ? v : 0.f;
      acc[t][r] = v; s1[r] += v; s2[r] += v * v;
    }
  }
#pragma unroll
  for (int m = 1; m <= 8; m <<= 1) {   // butterflies stay within each 16-lane half
#pragma unroll
    for (int r = 0; r < 8; ++r) {
      s1[r] += __shfl_xor(s1[r], m, 32);
      s2[r] += __shfl_xor(s2[r], m, 32);
    }
  }
#pragma unroll
  for (int r = 0; r < 8; ++r) {
    float mu = s1[r] * (1.f / 128.f);
    float var = s2[r] * (1.f / 128.f) - mu * mu;
    float rs = rsqrtf(var + LN_EPS);
    long gr = rbase + r + kh * 8;
    if (gr < N_NODES) {
#pragma unroll
      for (int t = 0; t < 8; ++t) {
        int col = t * 16 + n0;
        out[gr * 128 + col] = (acc[t][r] - mu) * rs * g[col] + bb[col];
      }
    }
  }
}

// ----------------------------- Kernel 6: graph mean pool --------------------------------
__global__ void pool_kernel(const float* __restrict__ x2, const int* __restrict__ batch,
                            float* __restrict__ sums, float* __restrict__ cnt) {
  int node = blockIdx.x, col = threadIdx.x;
  int b = batch[node];
  atomicAdd(&sums[b * 128 + col], x2[(long)node * 128 + col]);
  if (col == 0) atomicAdd(&cnt[b], 1.0f);
}

// ----------------------------- Kernel 7: MLP head (tiny) --------------------------------
__global__ void head_kernel(const float* __restrict__ sums, const float* __restrict__ cnt,
                            const float* __restrict__ u, const float* __restrict__ fw1,
                            const float* __restrict__ fb1, const float* __restrict__ fw2,
                            const float* __restrict__ fb2, const float* __restrict__ g,
                            const float* __restrict__ bb, float* __restrict__ out) {
  int i = threadIdx.x;
  if (i >= N_GRAPHS) return;
  float c = cnt[i]; if (c < 1.f) c = 1.f;
  float xf[160];
  for (int j = 0; j < 128; ++j) xf[j] = sums[i * 128 + j] / c;
  for (int j = 0; j < 32; ++j) xf[128 + j] = u[i * 32 + j];
  float y[32], s1 = 0.f, s2 = 0.f;
  for (int j = 0; j < 32; ++j) {
    float a = fb1[j];
    for (int k = 0; k < 160; ++k) a += xf[k] * fw1[k * 32 + j];
    a = a > 0.f ? a : 0.f;
    y[j] = a; s1 += a; s2 += a * a;
  }
  float mu = s1 / 32.f, var = s2 / 32.f - mu * mu, rs = rsqrtf(var + LN_EPS);
  float o = fb2[0];
  for (int j = 0; j < 32; ++j) o += ((y[j] - mu) * rs * g[j] + bb[j]) * fw2[j];
  out[i] = o;
}

extern "C" void kernel_launch(void* const* d_in, const int* in_sizes, int n_in,
                              void* d_out, int out_size, void* d_ws, size_t ws_size,
                              hipStream_t stream) {
  (void)in_sizes; (void)n_in; (void)out_size; (void)ws_size;
  const float* x         = (const float*)d_in[0];
  const float* edge_attr = (const float*)d_in[1];
  const float* u         = (const float*)d_in[2];
  const int*   eidx      = (const int*)d_in[3];
  const int*   batch     = (const int*)d_in[4];
  const float* em1_w1 = (const float*)d_in[5];  const float* em1_b1 = (const float*)d_in[6];
  const float* em1_w2 = (const float*)d_in[7];  const float* em1_b2 = (const float*)d_in[8];
  const float* em2_w1 = (const float*)d_in[9];  const float* em2_b1 = (const float*)d_in[10];
  const float* em2_w2 = (const float*)d_in[11]; const float* em2_b2 = (const float*)d_in[12];
  const float* c1_lin_w = (const float*)d_in[13]; const float* c1_lin_b = (const float*)d_in[14];
  const float* c1_w1 = (const float*)d_in[15];  const float* c1_b1 = (const float*)d_in[16];
  const float* c1_w2 = (const float*)d_in[17];  const float* c1_b2 = (const float*)d_in[18];
  const float* c2_lin_w = (const float*)d_in[19]; const float* c2_lin_b = (const float*)d_in[20];
  const float* c2_w1 = (const float*)d_in[21];  const float* c2_b1 = (const float*)d_in[22];
  const float* c2_w2 = (const float*)d_in[23];  const float* c2_b2 = (const float*)d_in[24];
  const float* lin1_w = (const float*)d_in[25]; const float* lin1_b = (const float*)d_in[26];
  const float* lin2_w = (const float*)d_in[27]; const float* lin2_b = (const float*)d_in[28];
  const float* fc1_w = (const float*)d_in[29];  const float* fc1_b = (const float*)d_in[30];
  const float* fc2_w = (const float*)d_in[31];  const float* fc2_b = (const float*)d_in[32];
  const float* ln1_g = (const float*)d_in[33];  const float* ln1_b = (const float*)d_in[34];
  const float* ln2_g = (const float*)d_in[35];  const float* ln2_b = (const float*)d_in[36];
  const float* ln3_g = (const float*)d_in[37];  const float* ln3_b = (const float*)d_in[38];
  const int* src = eidx;
  const int* dst = eidx + N_EDGES;

  char* w = (char*)d_ws;
  bf16*  ea   = (bf16*)w;  w += (size_t)N_EDGES * 128 * 2;        // 131 MB (reused layer 2)
  float* aggr = (float*)w; w += (size_t)3 * N_NODES * 128 * 4;    // 30.7 MB
  bf16*  t1   = (bf16*)w;  w += (size_t)3 * N_NODES * 128 * 2;    // 15.4 MB
  bf16*  xcat = (bf16*)w;  w += (size_t)N_NODES * 384 * 2;        // 15.4 MB
  float* x1   = (float*)w; w += (size_t)N_NODES * 128 * 4;        // 10.2 MB
  float* x2   = (float*)w; w += (size_t)N_NODES * 128 * 4;        // 10.2 MB
  float* sums = (float*)w; w += (size_t)N_GRAPHS * 128 * 4;
  float* cnt  = (float*)w; w += (size_t)N_GRAPHS * 4;

  dim3 blk(128);
  int EB = N_EDGES / 256;          // 2000 blocks, 4 tiles of 64 edges each
  int NB = (N_NODES + 63) / 64;    // 313

  // ---- layer 1 ----
  edge_mlp_kernel<<<EB, blk, 0, stream>>>(edge_attr, em1_w1, em1_b1, em1_w2, em1_b2, ea);
  hipMemsetAsync(aggr, 0, (size_t)3 * N_NODES * 128 * sizeof(float), stream);
  branch_edge_kernel<<<dim3(EB, 3), blk, 0, stream>>>(ea, c1_lin_w, c1_lin_b, x, src, dst, aggr);
  node_mlp1_kernel<<<dim3(NB, 3), blk, 0, stream>>>(x, aggr, c1_w1, c1_b1, t1);
  node_mlp2_kernel<<<dim3(NB, 3), blk, 0, stream>>>(t1, c1_w2, c1_b2, xcat);
  lin_ln_kernel<<<NB, blk, 0, stream>>>(xcat, lin1_w, lin1_b, ln1_g, ln1_b, x1);

  // ---- layer 2 ----
  edge_mlp_kernel<<<EB, blk, 0, stream>>>(edge_attr, em2_w1, em2_b1, em2_w2, em2_b2, ea);
  hipMemsetAsync(aggr, 0, (size_t)3 * N_NODES * 128 * sizeof(float), stream);
  branch_edge_kernel<<<dim3(EB, 3), blk, 0, stream>>>(ea, c2_lin_w, c2_lin_b, x1, src, dst, aggr);
  node_mlp1_kernel<<<dim3(NB, 3), blk, 0, stream>>>(x1, aggr, c2_w1, c2_b1, t1);
  node_mlp2_kernel<<<dim3(NB, 3), blk, 0, stream>>>(t1, c2_w2, c2_b2, xcat);
  lin_ln_kernel<<<NB, blk, 0, stream>>>(xcat, lin2_w, lin2_b, ln2_g, ln2_b, x2);

  // ---- readout ----
  hipMemsetAsync(sums, 0, (size_t)N_GRAPHS * 128 * sizeof(float), stream);
  hipMemsetAsync(cnt, 0, (size_t)N_GRAPHS * sizeof(float), stream);
  pool_kernel<<<N_NODES, 128, 0, stream>>>(x2, batch, sums, cnt);
  head_kernel<<<1, 64, 0, stream>>>(sums, cnt, u, fc1_w, fc1_b, fc2_w, fc2_b,
                                    ln3_g, ln3_b, (float*)d_out);
}
